// SeqAttention_24584392803106
// MI455X (gfx1250) — matrix operand order repaired and checked by reading, not compiled, not run on hardware
//
#include <hip/hip_runtime.h>
#include <hip/hip_bf16.h>

// Problem constants (match reference)
#define BB   4
#define NN   1024
#define CC   1024
#define HH   16
#define DD   64
#define MM   (BB * NN)   // 4096 rows of x
#define NQKV (3 * CC)    // 3072

typedef __attribute__((ext_vector_type(16))) __bf16 v16bf;
typedef __attribute__((ext_vector_type(8)))  float  v8f;

union Frag16 { v16bf v; uint4 q[2]; };

__device__ __forceinline__ v8f v8f_zero() {
  v8f x = {0.f, 0.f, 0.f, 0.f, 0.f, 0.f, 0.f, 0.f};
  return x;
}

__device__ __forceinline__ v8f wmma_bf16(v16bf a, v16bf b, v8f c) {
  return __builtin_amdgcn_wmma_f32_16x16x32_bf16(
      /*neg_a=*/false, a, /*neg_b=*/false, b,
      /*c_mod=*/(short)0, c, /*reuse_a=*/false, /*reuse_b=*/false);
}

__device__ __forceinline__ unsigned short f2bf_bits(float f) {
  union { __bf16 b; unsigned short u; } cv;
  cv.b = (__bf16)f;
  return cv.u;
}

// A fragment 16x32 bf16 (M x K), row-major source with leading dim ld.
// lane l: row = l&15; halves 0-7 -> K = k0 + 8*(l>>4) .. +7 ; halves 8-15 -> +16.
__device__ __forceinline__ v16bf load_frag_a(const __bf16* base, int ld,
                                             int row0, int k0, int lane) {
  const int r    = lane & 15;
  const int koff = (lane >> 4) << 3;
  const __bf16* p = base + (size_t)(row0 + r) * ld + k0 + koff;
  Frag16 f;
  f.q[0] = *(const uint4*)(p);
  f.q[1] = *(const uint4*)(p + 16);
  return f.v;
}

// B fragment 32x16 bf16 (K x N), where memory holds rows over N with K contiguous
// (i.e. src[n][k], leading dim ld).  lane l: col = l&15; halves 0-7 ->
// K = k0 + 8*(l>>4) .. +7 ; halves 8-15 -> +16.
__device__ __forceinline__ v16bf load_frag_b(const __bf16* base, int ld,
                                             int col0, int k0, int lane) {
  const int c    = lane & 15;
  const int koff = (lane >> 4) << 3;
  const __bf16* p = base + (size_t)(col0 + c) * ld + k0 + koff;
  Frag16 f;
  f.q[0] = *(const uint4*)(p);
  f.q[1] = *(const uint4*)(p + 16);
  return f.v;
}

// ---------------------------------------------------------------------------
// f32 -> bf16 conversion
// ---------------------------------------------------------------------------
__global__ __launch_bounds__(256) void cvt_bf16_kernel(const float* __restrict__ in,
                                                       __bf16* __restrict__ out, int n) {
  int i = blockIdx.x * blockDim.x + threadIdx.x;
  if (i < n) out[i] = (__bf16)in[i];
}

// ---------------------------------------------------------------------------
// QKV GEMM: qkv = x @ Wqkv^T ; scatter into Q (scaled), K, V^T  (all bf16)
// Wave tile 64x64; 8 waves/block.
// ---------------------------------------------------------------------------
__global__ __launch_bounds__(256) void qkv_gemm_kernel(
    const __bf16* __restrict__ xbf, const __bf16* __restrict__ wbf,
    __bf16* __restrict__ Qb, __bf16* __restrict__ Kb, __bf16* __restrict__ Vt) {
  const int lane  = threadIdx.x & 31;
  const int wave  = threadIdx.x >> 5;
  const int wtile = blockIdx.x * 8 + wave;          // 0 .. 64*48-1
  const int tm = (wtile % (MM / 64)) * 64;
  const int tn = (wtile / (MM / 64)) * 64;

  v8f acc[4][4];
  for (int i = 0; i < 4; i++)
    for (int j = 0; j < 4; j++) acc[i][j] = v8f_zero();

  for (int k0 = 0; k0 < CC; k0 += 32) {
    v16bf a[4], b[4];
#pragma unroll
    for (int i = 0; i < 4; i++) a[i] = load_frag_a(xbf, CC, tm + 16 * i, k0, lane);
#pragma unroll
    for (int j = 0; j < 4; j++) b[j] = load_frag_b(wbf, CC, tn + 16 * j, k0, lane);
#pragma unroll
    for (int i = 0; i < 4; i++)
#pragma unroll
      for (int j = 0; j < 4; j++)
        acc[i][j] = wmma_bf16(a[i], b[j], acc[i][j]);
  }

  // Epilogue: tn spans one 64-wide (s,h) block since tn % 64 == 0.
  const int c = lane & 15, hf = lane >> 4;
  const int s    = tn / CC;            // 0:Q 1:K 2:V
  const int hcol = (tn % CC) / DD;
#pragma unroll
  for (int i = 0; i < 4; i++) {
#pragma unroll
    for (int j = 0; j < 4; j++) {
      const int d = 16 * j + c;
#pragma unroll
      for (int g = 0; g < 8; g++) {
        const int m  = tm + 16 * i + g + 8 * hf;
        const int bb = m >> 10;
        const int n  = m & (NN - 1);
        const size_t hd = (size_t)bb * HH + hcol;
        const float v = acc[i][j][g];
        if (s == 0)      Qb[(hd * NN + n) * DD + d] = (__bf16)(v * 0.125f);
        else if (s == 1) Kb[(hd * NN + n) * DD + d] = (__bf16)v;
        else             Vt[(hd * DD + d) * NN + n] = (__bf16)v;
      }
    }
  }
}

// ---------------------------------------------------------------------------
// Flash attention: one wave per 16-row Q tile per head; online softmax.
// ---------------------------------------------------------------------------
__global__ __launch_bounds__(256) void attn_kernel(
    const __bf16* __restrict__ Qb, const __bf16* __restrict__ Kb,
    const __bf16* __restrict__ Vt, const float* __restrict__ mask,
    const unsigned char* __restrict__ lmask, __bf16* __restrict__ obf) {
  __shared__ __align__(16) unsigned short pbuf[8][16][40];  // per-wave 16x32 (+pad)

  const int lane = threadIdx.x & 31;
  const int wave = threadIdx.x >> 5;
  const int wg   = blockIdx.x * 8 + wave;  // 0 .. B*H*(N/16)-1 = 4095
  const int qt = wg & 63;                  // q tile in head
  const int bh = wg >> 6;                  // 0..63
  const int b_ = bh / HH, h = bh % HH;
  const int q0 = qt * 16;
  const int c = lane & 15, hf = lane >> 4;

  const __bf16* Qh = Qb + (size_t)bh * NN * DD;
  const __bf16* Kh = Kb + (size_t)bh * NN * DD;
  const __bf16* Vh = Vt + (size_t)bh * DD * NN;
  const float* maskb = mask + (size_t)b_ * NN;
  const unsigned char* lmb = lmask + (size_t)b_ * NN;

  const v16bf qf0 = load_frag_a(Qh, DD, q0, 0, lane);
  const v16bf qf1 = load_frag_a(Qh, DD, q0, 32, lane);

  float mq[8], badq[8];
#pragma unroll
  for (int g = 0; g < 8; g++) {
    const int q = q0 + g + 8 * hf;
    mq[g]   = maskb[q];
    badq[g] = lmb[q] ? 0.f : 1.f;
  }

  float m_i[8], l_i[8];
  v8f o[4];
#pragma unroll
  for (int g = 0; g < 8; g++) { m_i[g] = -3.0e38f; l_i[g] = 0.f; }
#pragma unroll
  for (int dc = 0; dc < 4; dc++) o[dc] = v8f_zero();

  unsigned short(*pb)[40] = pbuf[wave];

  for (int k0 = 0; k0 < NN; k0 += 32) {
    // ---- S = (Q*scale) K^T  for 32 key positions (two 16x16 tiles) ----
    const v16bf kb00 = load_frag_b(Kh, DD, k0,      0,  lane);
    const v16bf kb01 = load_frag_b(Kh, DD, k0,      32, lane);
    const v16bf kb10 = load_frag_b(Kh, DD, k0 + 16, 0,  lane);
    const v16bf kb11 = load_frag_b(Kh, DD, k0 + 16, 32, lane);
    v8f s0 = v8f_zero(), s1 = v8f_zero();
    s0 = wmma_bf16(qf0, kb00, s0);
    s0 = wmma_bf16(qf1, kb01, s0);
    s1 = wmma_bf16(qf0, kb10, s1);
    s1 = wmma_bf16(qf1, kb11, s1);

    // ---- additive bias ----
    const int kA = k0 + c, kB = k0 + 16 + c;
    const float mkA = maskb[kA], mkB = maskb[kB];
    const float badA = lmb[kA] ? 0.f : 1.f;
    const float badB = lmb[kB] ? 0.f : 1.f;
#pragma unroll
    for (int g = 0; g < 8; g++) {
      const float pA = (badq[g] + badA) > 0.f ? 1e9f : 0.f;
      const float pB = (badq[g] + badB) > 0.f ? 1e9f : 0.f;
      s0[g] += mq[g] + mkA - pA;
      s1[g] += mq[g] + mkB - pB;
    }

    // ---- online softmax (row reductions across 16-lane halves) ----
    float fac[8], p0[8], p1[8];
#pragma unroll
    for (int g = 0; g < 8; g++) {
      float t = fmaxf(s0[g], s1[g]);
      for (int off = 8; off; off >>= 1) t = fmaxf(t, __shfl_xor(t, off, 16));
      const float mnew = fmaxf(m_i[g], t);
      fac[g] = __expf(m_i[g] - mnew);
      p0[g]  = __expf(s0[g] - mnew);
      p1[g]  = __expf(s1[g] - mnew);
      float rs = p0[g] + p1[g];
      for (int off = 8; off; off >>= 1) rs += __shfl_xor(rs, off, 16);
      l_i[g] = l_i[g] * fac[g] + rs;
      m_i[g] = mnew;
    }
#pragma unroll
    for (int dc = 0; dc < 4; dc++)
#pragma unroll
      for (int g = 0; g < 8; g++) o[dc][g] *= fac[g];

    // ---- transpose P (C/D layout -> A layout) through per-wave LDS ----
#pragma unroll
    for (int g = 0; g < 8; g++) {
      const int r = g + 8 * hf;
      pb[r][c]      = f2bf_bits(p0[g]);
      pb[r][c + 16] = f2bf_bits(p1[g]);
    }
    Frag16 pf;
    {
      const int r    = lane & 15;
      const int koff = (lane >> 4) << 3;
      pf.q[0] = *(const uint4*)&pb[r][koff];
      pf.q[1] = *(const uint4*)&pb[r][koff + 16];
    }

    // ---- O += P V  (V stored transposed: contiguous along keys) ----
#pragma unroll
    for (int dc = 0; dc < 4; dc++) {
      const v16bf vb = load_frag_b(Vh, NN, dc * 16, k0, lane);
      o[dc] = wmma_bf16(pf.v, vb, o[dc]);
    }
  }

  // ---- normalize and store (B,N,H*D) as bf16 for the proj GEMM ----
#pragma unroll
  for (int g = 0; g < 8; g++) {
    const float inv = 1.f / l_i[g];
#pragma unroll
    for (int dc = 0; dc < 4; dc++) o[dc][g] *= inv;
  }
#pragma unroll
  for (int dc = 0; dc < 4; dc++) {
#pragma unroll
    for (int g = 0; g < 8; g++) {
      const int q   = q0 + g + 8 * hf;
      const int col = h * DD + dc * 16 + c;
      obf[((size_t)b_ * NN + q) * CC + col] = (__bf16)o[dc][g];
    }
  }
}

// ---------------------------------------------------------------------------
// Proj GEMM: out = attn_out @ Wproj^T + bproj   (f32 output)
// ---------------------------------------------------------------------------
__global__ __launch_bounds__(256) void proj_gemm_kernel(
    const __bf16* __restrict__ obf, const __bf16* __restrict__ wbf,
    const float* __restrict__ bias, float* __restrict__ out) {
  const int lane  = threadIdx.x & 31;
  const int wave  = threadIdx.x >> 5;
  const int wtile = blockIdx.x * 8 + wave;          // 0 .. 64*16-1
  const int tm = (wtile % (MM / 64)) * 64;
  const int tn = (wtile / (MM / 64)) * 64;

  v8f acc[4][4];
  for (int i = 0; i < 4; i++)
    for (int j = 0; j < 4; j++) acc[i][j] = v8f_zero();

  for (int k0 = 0; k0 < CC; k0 += 32) {
    v16bf a[4], b[4];
#pragma unroll
    for (int i = 0; i < 4; i++) a[i] = load_frag_a(obf, CC, tm + 16 * i, k0, lane);
#pragma unroll
    for (int j = 0; j < 4; j++) b[j] = load_frag_b(wbf, CC, tn + 16 * j, k0, lane);
#pragma unroll
    for (int i = 0; i < 4; i++)
#pragma unroll
      for (int j = 0; j < 4; j++)
        acc[i][j] = wmma_bf16(a[i], b[j], acc[i][j]);
  }

  const int c = lane & 15, hf = lane >> 4;
#pragma unroll
  for (int j = 0; j < 4; j++) {
    const int n  = tn + 16 * j + c;
    const float bv = bias[n];
#pragma unroll
    for (int i = 0; i < 4; i++) {
#pragma unroll
      for (int g = 0; g < 8; g++) {
        const int m = tm + 16 * i + g + 8 * hf;
        out[(size_t)m * CC + n] = acc[i][j][g] + bv;
      }
    }
  }
}

// ---------------------------------------------------------------------------
// Host launcher
// ---------------------------------------------------------------------------
extern "C" void kernel_launch(void* const* d_in, const int* in_sizes, int n_in,
                              void* d_out, int out_size, void* d_ws, size_t ws_size,
                              hipStream_t stream) {
  const float* x      = (const float*)d_in[0];          // (B,N,C)
  const float* mask   = (const float*)d_in[1];          // (B,N)
  // d_in[2] = lens (unused by the reference math)
  const unsigned char* lmask = (const unsigned char*)d_in[3];  // (B,N) bool
  const float* Wqkv   = (const float*)d_in[4];          // (3C,C)
  const float* Wproj  = (const float*)d_in[5];          // (C,C)
  const float* bproj  = (const float*)d_in[6];          // (C,)
  float* out = (float*)d_out;

  // Workspace carve-up (bf16 elements)
  __bf16* ws = (__bf16*)d_ws;
  __bf16* xbf    = ws;                      // 4096*1024
  __bf16* wqkvb  = xbf   + (size_t)MM * CC;       // 3072*1024
  __bf16* wprojb = wqkvb + (size_t)NQKV * CC;     // 1024*1024
  __bf16* Qb     = wprojb + (size_t)CC * CC;      // B*H*N*D
  __bf16* Kb     = Qb    + (size_t)MM * CC;
  __bf16* Vt     = Kb    + (size_t)MM * CC;
  __bf16* obf    = Vt    + (size_t)MM * CC;       // (B,N,C)

  // 1) f32 -> bf16 conversions
  {
    int n = MM * CC;
    cvt_bf16_kernel<<<(n + 255) / 256, 256, 0, stream>>>(x, xbf, n);
    n = NQKV * CC;
    cvt_bf16_kernel<<<(n + 255) / 256, 256, 0, stream>>>(Wqkv, wqkvb, n);
    n = CC * CC;
    cvt_bf16_kernel<<<(n + 255) / 256, 256, 0, stream>>>(Wproj, wprojb, n);
  }

  // 2) QKV projection: 64x64 wave tiles over 4096x3072 -> 3072 tiles / 8 waves
  qkv_gemm_kernel<<<(MM / 64) * (NQKV / 64) / 8, 256, 0, stream>>>(xbf, wqkvb, Qb, Kb, Vt);

  // 3) Attention: B*H*(N/16) = 4096 wave tiles / 8 waves per block
  attn_kernel<<<BB * HH * (NN / 16) / 8, 256, 0, stream>>>(Qb, Kb, Vt, mask, lmask, obf);

  // 4) Output projection: 64x64 wave tiles over 4096x1024 -> 1024 tiles / 8 waves
  proj_gemm_kernel<<<(MM / 64) * (CC / 64) / 8, 256, 0, stream>>>(obf, wprojb, bproj, out);
}
